// SVHPerm_20478404067505
// MI455X (gfx1250) — matrix operand order repaired and checked by reading, not compiled, not run on hardware
//
#include <hip/hip_runtime.h>
#include <math.h>

// ---------------------------------------------------------------------------
// SVHPerm on MI455X (gfx1250).
// Key idea: the 550MB partition-basis tensors are 0/1 equality masks; every
// equivariant linear map has a closed form in pooled stats (row/col sums,
// diag, trace, total).  We never read the basis inputs.  Each hidden layer
// collapses to two 64x64 x 64x2304 GEMMs (v_wmma_f32_16x16x32_bf16) plus
// tiny vector terms.  Final eigenvector via Gershgorin shift + repeated
// matrix squaring (v_wmma_f32_16x16x4_f32, 128^3 GEMM per squaring).
// ---------------------------------------------------------------------------

typedef __bf16 bf16;
typedef __attribute__((ext_vector_type(16))) __bf16 v16bf;
typedef __attribute__((ext_vector_type(8)))  float  v8f;
typedef __attribute__((ext_vector_type(2)))  float  v2f;

__device__ __forceinline__ float leaky(float v){ return v > 0.f ? v : 0.01f*v; }

// ---------------------------------------------------------------------------
// X0 = S S^T  (48x48 f32) + its pooled stats (symmetric: r==q).
// ---------------------------------------------------------------------------
__global__ void k_x0stats(const float* __restrict__ S, float* __restrict__ X0,
                          float* __restrict__ r0, float* __restrict__ g0,
                          float* __restrict__ TS0){
  __shared__ float sS[48*128];
  __shared__ float sX[48*48];
  __shared__ float sr[48];
  int t = threadIdx.x;
  for (int i = t; i < 48*128; i += 256) sS[i] = S[i];
  __syncthreads();
  for (int idx = t; idx < 2304; idx += 256){
    int c = idx/48, d = idx%48;
    float a = 0.f;
    for (int k = 0; k < 128; ++k) a += sS[c*128+k]*sS[d*128+k];
    sX[idx] = a; X0[idx] = a;
  }
  __syncthreads();
  if (t < 48){
    float a = 0.f;
    for (int d = 0; d < 48; ++d) a += sX[t*48+d];
    sr[t] = a; r0[t] = a; g0[t] = sX[t*49];
  }
  __syncthreads();
  if (t == 0){
    float T = 0.f, Sm = 0.f;
    for (int c = 0; c < 48; ++c){ T += sX[c*49]; Sm += sr[c]; }
    TS0[0] = T; TS0[1] = Sm;
  }
}

// ---------------------------------------------------------------------------
// Layer 0: 11 basis_sn orbit sums in closed form (X symmetric). One block per
// output channel o.  Writes bf16 activations.
// ---------------------------------------------------------------------------
__global__ void k_layer0(const float* __restrict__ w0, const float* __restrict__ b0,
                         const float* __restrict__ X0, const float* __restrict__ r0,
                         const float* __restrict__ g0, const float* __restrict__ TS0,
                         bf16* __restrict__ Xout){
  int o = blockIdx.x, t = threadIdx.x;
  const float* w = w0 + o*11;
  float w0_=w[0],w1_=w[1],w2_=w[2],w3_=w[3],w4_=w[4],w5_=w[5],w6_=w[6],
        w7_=w[7],w8_=w[8],w9_=w[9],w10_=w[10];
  // off-diagonal coefficients
  float aX  = 2.f*(w0_-w2_-w4_+w10_);
  float agc = 2.f*w0_-2.f*w2_-w5_+w7_;
  float agd = 2.f*w0_-2.f*w4_-w5_+w9_;
  float arc = -2.f*w0_+2.f*w2_;
  float ard = -2.f*w0_+2.f*w4_;
  float aT  = -w0_+w5_;
  float aS  =  w0_;
  // diagonal coefficients
  float dS = w1_, dT = -w1_+w6_, dr = -2.f*w1_+2.f*w3_, dg = 2.f*w1_-2.f*w3_-w6_+w8_;
  float boff = b0[o*2+0], bdia = b0[o*2+1];
  float T0 = TS0[0], S0 = TS0[1];
  for (int idx = t; idx < 2304; idx += 256){
    int c = idx/48, d = idx%48;
    float v;
    if (c == d) v = dS*S0 + dT*T0 + dr*r0[c] + dg*g0[c] + bdia;
    else        v = aX*X0[idx] + agc*g0[c] + agd*g0[d] + arc*r0[c] + ard*r0[d]
                    + aT*T0 + aS*S0 + boff;
    Xout[o*2304 + idx] = (bf16)leaky(v);
  }
}

// ---------------------------------------------------------------------------
// Pooled stats of a 64x48x48 bf16 activation tensor. One block per channel.
// ---------------------------------------------------------------------------
__global__ void k_stats(const bf16* __restrict__ X, float* __restrict__ r,
                        float* __restrict__ q, float* __restrict__ g,
                        float* __restrict__ Tv, float* __restrict__ Sv){
  int i = blockIdx.x, t = threadIdx.x;
  __shared__ float sX[2304];
  __shared__ float sr[48];
  for (int idx = t; idx < 2304; idx += 256) sX[idx] = (float)X[i*2304+idx];
  __syncthreads();
  if (t < 48){
    float a = 0.f;
    for (int d = 0; d < 48; ++d) a += sX[t*48+d];
    r[i*48+t] = a; sr[t] = a;
  } else if (t >= 64 && t < 112){
    int d = t-64; float a = 0.f;
    for (int c = 0; c < 48; ++c) a += sX[c*48+d];
    q[i*48+d] = a;
  } else if (t >= 128 && t < 176){
    int c = t-128; g[i*48+c] = sX[c*49];
  }
  __syncthreads();
  if (t == 0){
    float T = 0.f, Sm = 0.f;
    for (int c = 0; c < 48; ++c){ T += sX[c*49]; Sm += sr[c]; }
    Tv[i] = T; Sv[i] = Sm;
  }
}

// ---------------------------------------------------------------------------
// Derived GEMM coefficient matrices for hidden layer l:
//  cX[o,i]  = W11-W10-W4+W0   (coefficient of X_i[c,d])
//  cXT[o,i] = W14-W13-W2+W0   (coefficient of X_i[d,c])
// ---------------------------------------------------------------------------
__global__ void k_coef(const float* __restrict__ wh, int l,
                       bf16* __restrict__ cX, bf16* __restrict__ cXT){
  int id = blockIdx.x*256 + threadIdx.x;        // id = o*64+i
  const float* W = wh + (size_t)(l*4096 + id)*15;
  __builtin_prefetch(W, 0, 1);
  cX[id]  = (bf16)(W[11]-W[10]-W[4]+W[0]);
  cXT[id] = (bf16)(W[14]-W[13]-W[2]+W[0]);
}

// ---------------------------------------------------------------------------
// Vector/scalar terms of a hidden layer (closed-form partition pooling).
// One block per output channel o, 192 threads.
// ---------------------------------------------------------------------------
__global__ void k_vec(const float* __restrict__ wh, const float* __restrict__ bh, int l,
                      const float* __restrict__ r, const float* __restrict__ q,
                      const float* __restrict__ g, const float* __restrict__ Tv,
                      const float* __restrict__ Sv,
                      float* __restrict__ Voffc, float* __restrict__ Voffd,
                      float* __restrict__ Vdiag, float* __restrict__ soff,
                      float* __restrict__ sdia){
  int o = blockIdx.x, t = threadIdx.x;
  __shared__ float sW[960];                      // wh[l][o][0..63][0..14]
  const float* base = wh + (size_t)(l*4096 + o*64)*15;
  for (int idx = t; idx < 960; idx += 192) sW[idx] = base[idx];
  __syncthreads();
  if (t < 48){
    int c = t; float a = 0.f;
    for (int i = 0; i < 64; ++i){
      const float* W = &sW[i*15];
      float cgc = W[7]-W[10]-W[2]-W[5]+2.f*W[0];
      float crc = W[10]-W[0];
      float cqc = W[2]-W[0];
      a += cgc*g[i*48+c] + crc*r[i*48+c] + cqc*q[i*48+c];
    }
    Voffc[o*48+c] = a;
  } else if (t < 96){
    int d = t-48; float a = 0.f;
    for (int i = 0; i < 64; ++i){
      const float* W = &sW[i*15];
      float cgd = W[9]-W[13]-W[4]-W[5]+2.f*W[0];
      float crd = W[13]-W[0];
      float cqd = W[4]-W[0];
      a += cgd*g[i*48+d] + crd*r[i*48+d] + cqd*q[i*48+d];
    }
    Voffd[o*48+d] = a;
  } else if (t < 144){
    int c = t-96; float a = 0.f;
    for (int i = 0; i < 64; ++i){
      const float* W = &sW[i*15];
      float dgc = W[8]-W[12]-W[3]-W[6]+2.f*W[1];
      float drc = W[12]-W[1];
      float dqc = W[3]-W[1];
      a += dgc*g[i*48+c] + drc*r[i*48+c] + dqc*q[i*48+c];
    }
    Vdiag[o*48+c] = a;
  } else if (t == 144){
    float a = 0.f;
    for (int i = 0; i < 64; ++i){
      const float* W = &sW[i*15];
      a += (W[5]-W[0])*Tv[i] + W[0]*Sv[i];
    }
    soff[o] = a + bh[(l*64+o)*2+0];
  } else if (t == 145){
    float a = 0.f;
    for (int i = 0; i < 64; ++i){
      const float* W = &sW[i*15];
      a += (W[6]-W[1])*Tv[i] + W[1]*Sv[i];
    }
    sdia[o] = a + bh[(l*64+o)*2+1];
  }
}

// ---------------------------------------------------------------------------
// Pass A: G2 = cXT (*) X, stored TRANSPOSED over (c,d): G2T[o][d*48+c].
// M=64 (4 waves x 16), N-tile = 16 columns of flattened n=c*48+d, K=64.
// ---------------------------------------------------------------------------
__global__ void k_gemmA(const bf16* __restrict__ Xin, const bf16* __restrict__ cXT,
                        float* __restrict__ G2T){
  __shared__ bf16 sXt[16*64];     // [nn][i]  (so 16 K-values are contiguous)
  __shared__ bf16 sA[64*64];
  int t = threadIdx.x, n0 = blockIdx.x*16;
  for (int idx = t; idx < 1024; idx += 128){
    int nn = idx & 15, i = idx >> 4;
    sXt[nn*64+i] = Xin[i*2304 + n0 + nn];
  }
  for (int idx = t; idx < 4096; idx += 128) sA[idx] = cXT[idx];
  __syncthreads();
  int wave = t >> 5, L = t & 31, m = L & 15, half = L >> 4;
  v8f acc;
  #pragma unroll
  for (int v = 0; v < 8; ++v) acc[v] = 0.f;
  #pragma unroll
  for (int ks = 0; ks < 2; ++ks){
    v16bf a, b;
    int kbA = ks*32 + half*16;
    #pragma unroll
    for (int tt = 0; tt < 16; ++tt){
      int K = kbA + tt;
      a[tt] = sA[(wave*16+m)*64 + K];
    }
    int kbB = ks*32 + half*16;
    #pragma unroll
    for (int tt = 0; tt < 16; ++tt) b[tt] = sXt[m*64 + kbB + tt];
    acc = __builtin_amdgcn_wmma_f32_16x16x32_bf16(false, a, false, b, (short)0, acc, false, false);
  }
  #pragma unroll
  for (int v = 0; v < 8; ++v){
    int o = wave*16 + v + 8*half;
    int n = n0 + m, c = n/48, d = n%48;
    G2T[o*2304 + d*48 + c] = acc[v];
  }
}

// ---------------------------------------------------------------------------
// Pass B: G1 = cX (*) X ; assemble layer output with transpose term, vector
// terms, bias, leaky-relu ; write next activations (bf16).
// ---------------------------------------------------------------------------
__global__ void k_gemmB(const bf16* __restrict__ Xin, const bf16* __restrict__ cX,
                        const float* __restrict__ G2T,
                        const float* __restrict__ Voffc, const float* __restrict__ Voffd,
                        const float* __restrict__ Vdiag, const float* __restrict__ soff,
                        const float* __restrict__ sdia, bf16* __restrict__ Xout){
  __shared__ bf16 sXt[16*64];
  __shared__ bf16 sA[64*64];
  int t = threadIdx.x, n0 = blockIdx.x*16;
  for (int idx = t; idx < 1024; idx += 128){
    int nn = idx & 15, i = idx >> 4;
    sXt[nn*64+i] = Xin[i*2304 + n0 + nn];
  }
  for (int idx = t; idx < 4096; idx += 128) sA[idx] = cX[idx];
  __syncthreads();
  int wave = t >> 5, L = t & 31, m = L & 15, half = L >> 4;
  v8f acc;
  #pragma unroll
  for (int v = 0; v < 8; ++v) acc[v] = 0.f;
  #pragma unroll
  for (int ks = 0; ks < 2; ++ks){
    v16bf a, b;
    int kbA = ks*32 + half*16;
    #pragma unroll
    for (int tt = 0; tt < 16; ++tt){
      int K = kbA + tt;
      a[tt] = sA[(wave*16+m)*64 + K];
    }
    int kbB = ks*32 + half*16;
    #pragma unroll
    for (int tt = 0; tt < 16; ++tt) b[tt] = sXt[m*64 + kbB + tt];
    acc = __builtin_amdgcn_wmma_f32_16x16x32_bf16(false, a, false, b, (short)0, acc, false, false);
  }
  #pragma unroll
  for (int v = 0; v < 8; ++v){
    int o = wave*16 + v + 8*half;
    int n = n0 + m, c = n/48, d = n%48;
    float val;
    if (c == d) val = Vdiag[o*48+c] + sdia[o];
    else        val = acc[v] + G2T[o*2304+n] + Voffc[o*48+c] + Voffd[o*48+d] + soff[o];
    Xout[o*2304 + n] = (bf16)leaky(val);
  }
}

// ---------------------------------------------------------------------------
// Pairs layer (transposed basis_sn orbits) + identity scalar lambda.
// ---------------------------------------------------------------------------
__global__ void k_pairs(const float* __restrict__ wp, const float* __restrict__ bp,
                        const float* __restrict__ wi, const float* __restrict__ bi,
                        const bf16* __restrict__ X,
                        const float* __restrict__ r, const float* __restrict__ q,
                        const float* __restrict__ g, const float* __restrict__ Tv,
                        const float* __restrict__ Sv,
                        float* __restrict__ P, float* __restrict__ lam){
  // combo slots: 0 pX, 1 pg, 2 pr, 3 pq, 4 pT, 5 pS, 6 pdg, 7 pdr, 8 pdq, 9 pdT, 10 pdS
  __shared__ float cf[11*64];
  __shared__ float idp[64];
  int t = threadIdx.x;
  if (t < 64){
    const float* w = wp + t*11;
    float w0=w[0],w1=w[1],w2=w[2],w3=w[3],w4=w[4],w5=w[5],w6=w[6],w7=w[7],w8=w[8],w9=w[9],w10=w[10];
    cf[0*64+t]  = w0 - w2 - w4 + w10;
    cf[1*64+t]  = 2.f*w0 - w1 - w2 + w3 - w4;
    cf[2*64+t]  = -w0 + w2;
    cf[3*64+t]  = -w0 + w4;
    cf[4*64+t]  = -w0 + w1;
    cf[5*64+t]  =  w0;
    cf[6*64+t]  = 2.f*w5 - w6 - w7 + w8 - w9;
    cf[7*64+t]  = -w5 + w7;
    cf[8*64+t]  = -w5 + w9;
    cf[9*64+t]  = -w5 + w6;
    cf[10*64+t] =  w5;
    idp[t] = wi[t*2+0]*(Sv[t]-Tv[t]) + wi[t*2+1]*Tv[t];
  }
  __syncthreads();
  if (t == 0){
    float a = 0.f;
    for (int i = 0; i < 64; ++i) a += idp[i];
    lam[0] = a + bi[0];
  }
  for (int idx = t; idx < 2304; idx += 256){
    int c = idx/48, d = idx%48;
    float acc = 0.f;
    if (c == d){
      for (int i = 0; i < 64; ++i)
        acc += cf[6*64+i]*g[i*48+c] + cf[7*64+i]*r[i*48+c] + cf[8*64+i]*q[i*48+c]
             + cf[9*64+i]*Tv[i] + cf[10*64+i]*Sv[i];
      acc += bp[1];
    } else {
      for (int i = 0; i < 64; ++i){
        float xs = (float)X[i*2304 + idx] + (float)X[i*2304 + d*48 + c];
        acc += cf[0*64+i]*xs
             + cf[1*64+i]*(g[i*48+c]+g[i*48+d])
             + cf[2*64+i]*(r[i*48+c]+r[i*48+d])
             + cf[3*64+i]*(q[i*48+c]+q[i*48+d])
             + cf[4*64+i]*Tv[i] + cf[5*64+i]*Sv[i];
      }
      acc += bp[0];
    }
    P[idx] = acc;
  }
}

// ---------------------------------------------------------------------------
// A = S^T P S + lambda*I, via two WMMA f32 16x16x4 GEMM phases in one block.
// ---------------------------------------------------------------------------
__global__ void k_buildA(const float* __restrict__ S, const float* __restrict__ P,
                         const float* __restrict__ lam, float* __restrict__ A){
  __shared__ float sP[48*48];      // 9216 B
  __shared__ float sS[48*128];     // 24576 B
  __shared__ float sM[48*128];     // 24576 B  (M1 = P S)
  int t = threadIdx.x;
  for (int i = t; i < 2304; i += 256) sP[i] = P[i];
  for (int i = t; i < 6144; i += 256) sS[i] = S[i];
  __syncthreads();
  int wave = t >> 5, L = t & 31, m = L & 15, half = L >> 4;
  // Phase 1: M1(48x128) = P(48x48) * S(48x128); 24 tiles, 3 per wave.
  for (int j = 0; j < 3; ++j){
    int tt = wave*3 + j, mt = tt/8, nt = tt%8;
    v8f acc;
    #pragma unroll
    for (int v = 0; v < 8; ++v) acc[v] = 0.f;
    for (int ks = 0; ks < 12; ++ks){
      v2f a, b;
      int k0 = ks*4 + half*2;
      a[0] = sP[(mt*16+m)*48 + k0];
      a[1] = sP[(mt*16+m)*48 + k0+1];
      b[0] = sS[(k0  )*128 + nt*16+m];
      b[1] = sS[(k0+1)*128 + nt*16+m];
      acc = __builtin_amdgcn_wmma_f32_16x16x4_f32(false, a, false, b, (short)0, acc, false, false);
    }
    #pragma unroll
    for (int v = 0; v < 8; ++v) sM[(mt*16 + v + 8*half)*128 + nt*16+m] = acc[v];
  }
  __syncthreads();
  float lv = lam[0];
  // Phase 2: A(128x128) = S^T(128x48) * M1(48x128); 64 tiles, 8 per wave.
  for (int j = 0; j < 8; ++j){
    int tt = wave*8 + j, mt = tt/8, nt = tt%8;
    v8f acc;
    #pragma unroll
    for (int v = 0; v < 8; ++v) acc[v] = 0.f;
    for (int ks = 0; ks < 12; ++ks){
      v2f a, b;
      int k0 = ks*4 + half*2;
      a[0] = sS[(k0  )*128 + mt*16+m];   // S^T[row][k] = S[k][row]
      a[1] = sS[(k0+1)*128 + mt*16+m];
      b[0] = sM[(k0  )*128 + nt*16+m];
      b[1] = sM[(k0+1)*128 + nt*16+m];
      acc = __builtin_amdgcn_wmma_f32_16x16x4_f32(false, a, false, b, (short)0, acc, false, false);
    }
    #pragma unroll
    for (int v = 0; v < 8; ++v){
      int row = mt*16 + v + 8*half, col = nt*16 + m;
      A[row*128 + col] = acc[v] + (row == col ? lv : 0.f);
    }
  }
}

// ---------------------------------------------------------------------------
// Gershgorin shift (B0 = A + sigma*I, guarantees PSD and preserves the top
// algebraic eigenvector) + init trace scale for the squaring chain.
// ---------------------------------------------------------------------------
__global__ void k_prep(float* __restrict__ A, float* __restrict__ scacc){
  __shared__ float rs[128];
  __shared__ float sdg[128];
  __shared__ float sigma;
  int t = threadIdx.x;
  if (t < 128){
    float a = 0.f;
    for (int c = 0; c < 128; ++c) a += fabsf(A[t*128+c]);
    rs[t] = a;
  }
  __syncthreads();
  if (t == 0){
    float mx = 0.f;
    for (int i = 0; i < 128; ++i) mx = fmaxf(mx, rs[i]);
    sigma = mx;
  }
  __syncthreads();
  if (t < 128){
    float dv = A[t*128+t] + sigma;
    A[t*128+t] = dv;
    sdg[t] = dv;
  }
  __syncthreads();
  if (t == 0){
    float tr = 0.f;
    for (int i = 0; i < 128; ++i) tr += sdg[i];
    scacc[0] = (tr > 1e-30f ? tr : 1.f);
  }
  if (t >= 1 && t < 24) scacc[t] = 0.f;
}

// ---------------------------------------------------------------------------
// One squaring step: Bout = (Bin/trace_prev) * Bin ; accumulate trace(Bout).
// 64 tiles of 16x16, K=128, v_wmma_f32_16x16x4_f32. grid=16 x 128 threads.
// ---------------------------------------------------------------------------
__global__ void k_square(const float* __restrict__ Bin, float* __restrict__ Bout,
                         const float* __restrict__ scPrev, float* __restrict__ scNext){
  int t = threadIdx.x;
  int wave = t >> 5, L = t & 31, m = L & 15, half = L >> 4;
  int tile = blockIdx.x*4 + wave;
  int mt = tile >> 3, nt = tile & 7;
  float s = 1.f / scPrev[0];
  v8f acc;
  #pragma unroll
  for (int v = 0; v < 8; ++v) acc[v] = 0.f;
  for (int ks = 0; ks < 32; ++ks){
    v2f a, b;
    int k0 = ks*4 + half*2;
    a[0] = s * Bin[(mt*16+m)*128 + k0];
    a[1] = s * Bin[(mt*16+m)*128 + k0+1];
    b[0] = Bin[(k0  )*128 + nt*16+m];
    b[1] = Bin[(k0+1)*128 + nt*16+m];
    acc = __builtin_amdgcn_wmma_f32_16x16x4_f32(false, a, false, b, (short)0, acc, false, false);
  }
  #pragma unroll
  for (int v = 0; v < 8; ++v){
    int row = mt*16 + v + 8*half, col = nt*16 + m;
    Bout[row*128 + col] = acc[v];
    if (row == col) atomicAdd(scNext, acc[v]);
  }
}

// ---------------------------------------------------------------------------
// Extract dominant column of B^(2^K) (rank-1 limit mu*u*u^T), normalize,
// output S*u.
// ---------------------------------------------------------------------------
__global__ void k_extract(const float* __restrict__ B, const float* __restrict__ S,
                          float* __restrict__ out){
  __shared__ float sd[128];
  __shared__ float u[128];
  __shared__ int jbest;
  __shared__ float nrm;
  int t = threadIdx.x;
  if (t < 128) sd[t] = fabsf(B[t*128+t]);
  __syncthreads();
  if (t == 0){
    int jb = 0; float bv = -1.f;
    for (int j = 0; j < 128; ++j) if (sd[j] > bv){ bv = sd[j]; jb = j; }
    jbest = jb;
  }
  __syncthreads();
  if (t < 128) u[t] = B[t*128 + jbest];
  __syncthreads();
  if (t == 0){
    float s2 = 0.f;
    for (int k = 0; k < 128; ++k) s2 += u[k]*u[k];
    nrm = rsqrtf(fmaxf(s2, 1e-30f));
  }
  __syncthreads();
  if (t < 48){
    float a = 0.f;
    for (int k = 0; k < 128; ++k) a += S[t*128+k]*u[k];
    out[t] = a * nrm;
  }
}

// ---------------------------------------------------------------------------
extern "C" void kernel_launch(void* const* d_in, const int* in_sizes, int n_in,
                              void* d_out, int out_size, void* d_ws, size_t ws_size,
                              hipStream_t stream){
  const float* S  = (const float*)d_in[0];
  const float* w0 = (const float*)d_in[1];
  const float* b0 = (const float*)d_in[2];
  const float* wh = (const float*)d_in[3];
  const float* bh = (const float*)d_in[4];
  const float* wp = (const float*)d_in[5];
  const float* bp = (const float*)d_in[6];
  const float* wi = (const float*)d_in[7];
  const float* bi = (const float*)d_in[8];
  // basis_sn / basis_nn / bias_basis (d_in[9..11]) are structured 0/1 masks:
  // evaluated analytically, never read (saves ~550MB of HBM traffic).

  char* base = (char*)d_ws;
  bf16*  Xa    = (bf16*)(base + 0);          // 64*2304 bf16
  bf16*  Xb    = (bf16*)(base + 294912);
  float* G2T   = (float*)(base + 589824);    // 64*2304 f32
  float* r     = (float*)(base + 1179648);   // 64*48
  float* q     = (float*)(base + 1191936);
  float* g     = (float*)(base + 1204224);
  float* Tv    = (float*)(base + 1216512);   // 64
  float* Sv    = (float*)(base + 1216768);
  bf16*  cX    = (bf16*)(base + 1217024);    // 64*64 bf16
  bf16*  cXT   = (bf16*)(base + 1225216);
  float* Voffc = (float*)(base + 1233408);   // 64*48
  float* Voffd = (float*)(base + 1245696);
  float* Vdiag = (float*)(base + 1257984);
  float* soff  = (float*)(base + 1270272);   // 64
  float* sdia  = (float*)(base + 1270528);
  float* X0    = (float*)(base + 1270784);   // 48*48
  float* r0    = (float*)(base + 1280000);   // 48
  float* g0    = (float*)(base + 1280192);
  float* TS0   = (float*)(base + 1280384);   // 2
  float* P     = (float*)(base + 1280512);   // 48*48
  float* Ba    = (float*)(base + 1289728);   // 128*128
  float* Bb    = (float*)(base + 1355264);
  float* lam   = (float*)(base + 1420800);
  float* scacc = (float*)(base + 1420804);   // 24 floats

  // Input layer
  k_x0stats<<<1, 256, 0, stream>>>(S, X0, r0, g0, TS0);
  k_layer0<<<64, 256, 0, stream>>>(w0, b0, X0, r0, g0, TS0, Xa);
  k_stats<<<64, 256, 0, stream>>>(Xa, r, q, g, Tv, Sv);

  // Hidden layers (ping-pong Xa <-> Xb)
  bf16* Xin = Xa; bf16* Xout = Xb;
  for (int l = 0; l < 3; ++l){
    k_coef<<<16, 256, 0, stream>>>(wh, l, cX, cXT);
    k_vec<<<64, 192, 0, stream>>>(wh, bh, l, r, q, g, Tv, Sv,
                                  Voffc, Voffd, Vdiag, soff, sdia);
    k_gemmA<<<144, 128, 0, stream>>>(Xin, cXT, G2T);
    k_gemmB<<<144, 128, 0, stream>>>(Xin, cX, G2T, Voffc, Voffd, Vdiag, soff, sdia, Xout);
    k_stats<<<64, 256, 0, stream>>>(Xout, r, q, g, Tv, Sv);
    bf16* tmp = Xin; Xin = Xout; Xout = tmp;
  }
  // After 3 layers the final activations are in Xin (== Xb).
  k_pairs<<<1, 256, 0, stream>>>(wp, bp, wi, bi, Xin, r, q, g, Tv, Sv, P, lam);
  k_buildA<<<1, 256, 0, stream>>>(S, P, lam, Ba);
  k_prep<<<1, 256, 0, stream>>>(Ba, scacc);

  // 18 matrix squarings: effective power 2^18 -> dominant eigenvector.
  float* bin = Ba; float* bout = Bb;
  for (int it = 0; it < 18; ++it){
    k_square<<<16, 128, 0, stream>>>(bin, bout, scacc + it, scacc + it + 1);
    float* tmp = bin; bin = bout; bout = tmp;
  }
  // 18 iterations (even) -> result back in Ba (== bin after the swaps).
  k_extract<<<1, 256, 0, stream>>>(bin, S, (float*)d_out);
}
